// Linear_34067680592363
// MI455X (gfx1250) — compile-verified, matrix operand order checked by reading, not run on hardware
//
#include <hip/hip_runtime.h>

// ---------------------------------------------------------------------------
// NF4 linear on gfx1250 (MI455X):
//   pass 1a: x fp32 -> bf16  (ws)
//   pass 1b: packed NF4 codes + per-row scale -> W bf16 (ws)
//   pass 2 : tiled bf16 WMMA GEMM  out = x @ W^T + bias  (fp32 out)
// Compute-bound (AI ~1000 flop/byte >> ~100 machine balance). GEMM rides
// v_wmma_f32_16x16x32_bf16 with double-buffered LDS tiles filled by the
// gfx1250 async-to-LDS path (ASYNCcnt) so staging overlaps the matrix pipe.
// ---------------------------------------------------------------------------

typedef __bf16 bf16_t;
typedef __attribute__((ext_vector_type(8)))  __bf16 v8bf;
typedef __attribute__((ext_vector_type(16))) __bf16 v16bf;
typedef __attribute__((ext_vector_type(8)))  float  v8f;
typedef __attribute__((ext_vector_type(4)))  unsigned int u32x4;
typedef int v4i __attribute__((vector_size(16)));   // matches builtin pointee

#define IN_F   4096
#define OUT_F  4096
#define BM     128
#define BN     128
#define BK     32
#define LDT    40   // padded bf16 LDS row stride (80 B -> staggers banks)

#if defined(__has_builtin)
#  if __has_builtin(__builtin_amdgcn_global_load_async_to_lds_b128)
#    define HAS_ASYNC_LDS 1
#  else
#    define HAS_ASYNC_LDS 0
#  endif
#  if __has_builtin(__builtin_amdgcn_s_wait_asynccnt)
#    define HAS_WAIT_ASYNC_BUILTIN 1
#  else
#    define HAS_WAIT_ASYNC_BUILTIN 0
#  endif
#else
#  define HAS_ASYNC_LDS 0
#  define HAS_WAIT_ASYNC_BUILTIN 0
#endif

// Typed address-space casts matching the async builtin's signature:
//   (v4i addrspace(1)* src_global, v4i addrspace(3)* dst_lds, Ii off, Ii cpol)
#define ASG(p) ((__attribute__((address_space(1))) v4i*)(p))
#define ASL(p) ((__attribute__((address_space(3))) v4i*)(p))

__device__ const float NF4_CB[16] = {
    -1.0f, -0.6961928f, -0.52507305f, -0.3949175f, -0.28444138f, -0.18477343f,
    -0.09105004f, 0.0f, 0.0795803f, 0.1609302f, 0.2461123f, 0.33791524f,
    0.44070983f, 0.562617f, 0.72295684f, 1.0f};

// ---- pass 1a: fp32 -> bf16, 8 elements per thread --------------------------
__global__ __launch_bounds__(256) void nf4_convert_x(
    const float* __restrict__ x, bf16_t* __restrict__ xb) {
  const size_t t = (size_t)blockIdx.x * blockDim.x + threadIdx.x;
  const float4* src = (const float4*)x + t * 2;
  float4 a = src[0];
  float4 b = src[1];
  v8bf v;
  v[0] = (bf16_t)a.x; v[1] = (bf16_t)a.y; v[2] = (bf16_t)a.z; v[3] = (bf16_t)a.w;
  v[4] = (bf16_t)b.x; v[5] = (bf16_t)b.y; v[6] = (bf16_t)b.z; v[7] = (bf16_t)b.w;
  *((v8bf*)xb + t) = v;
}

// ---- pass 1b: NF4 nibbles -> bf16 weights, 4 packed ints per thread --------
__global__ __launch_bounds__(256) void nf4_dequant_w(
    const int* __restrict__ codes, const float* __restrict__ scale,
    bf16_t* __restrict__ wb) {
  __shared__ float cb[16];
  if (threadIdx.x < 16) cb[threadIdx.x] = NF4_CB[threadIdx.x];
  __syncthreads();

  const size_t t = (size_t)blockIdx.x * blockDim.x + threadIdx.x;
  const size_t j = t * 4;                 // packed-int index; 2 weights each
  const int o = (int)(j >> 11);           // row = (2*j)/IN_F, IN_F/2 = 2048
  const float s = scale[o];

  int4 c = *((const int4*)codes + t);
  int cc[4] = {c.x, c.y, c.z, c.w};
  v8bf v;
#pragma unroll
  for (int q = 0; q < 4; ++q) {
    int b = cc[q] & 0xFF;
    v[2 * q]     = (bf16_t)(cb[b & 15] * s);
    v[2 * q + 1] = (bf16_t)(cb[(b >> 4) & 15] * s);
  }
  *((v8bf*)wb + t) = v;
}

// ---- helpers ---------------------------------------------------------------
__device__ inline void wait_async0() {
#if HAS_WAIT_ASYNC_BUILTIN
  __builtin_amdgcn_s_wait_asynccnt(0);
#elif HAS_ASYNC_LDS
  asm volatile("s_wait_asynccnt 0" ::: "memory");
#endif
}

// Stage one 128x32 bf16 tile of A and B into LDS. 512 16-byte segments each;
// every thread owns 2 segments per operand.
__device__ inline void stage_tile(const bf16_t* __restrict__ X,
                                  const bf16_t* __restrict__ W,
                                  bf16_t* __restrict__ As,
                                  bf16_t* __restrict__ Bs,
                                  int bm, int bn, int k0, int tid) {
#pragma unroll
  for (int i = 0; i < 2; ++i) {
    int s   = tid + i * 256;
    int row = s >> 2;             // 4 segments per 32-elem row
    int kc  = (s & 3) * 8;        // bf16 offset within row
    const bf16_t* ga = X + (size_t)(bm + row) * IN_F + k0 + kc;
    const bf16_t* gb = W + (size_t)(bn + row) * IN_F + k0 + kc;
    bf16_t* la = As + row * LDT + kc;
    bf16_t* lb = Bs + row * LDT + kc;
#if HAS_ASYNC_LDS
    __builtin_amdgcn_global_load_async_to_lds_b128(ASG(ga), ASL(la), 0, 0);
    __builtin_amdgcn_global_load_async_to_lds_b128(ASG(gb), ASL(lb), 0, 0);
#else
    *(u32x4*)la = *(const u32x4*)ga;
    *(u32x4*)lb = *(const u32x4*)gb;
#endif
  }
}

// ---- LDS -> v16bf fragment per ISA 16-bit A/B layout -----------------------
// lane 0..15 : row = lane,    K = {0..7, 16..23}
// lane 16..31: row = lane-16, K = {8..15, 24..31}
__device__ inline v16bf load_frag(const bf16_t* __restrict__ p, int kh) {
  v8bf lo = *(const v8bf*)(p + kh);
  v8bf hi = *(const v8bf*)(p + 16 + kh);
  return __builtin_shufflevector(lo, hi, 0, 1, 2, 3, 4, 5, 6, 7,
                                 8, 9, 10, 11, 12, 13, 14, 15);
}

// ---- pass 2: bf16 WMMA GEMM, double-buffered async staging -----------------
__global__ __launch_bounds__(256) void nf4_gemm(
    const bf16_t* __restrict__ X,   // [M][IN_F]  row-major
    const bf16_t* __restrict__ W,   // [OUT_F][IN_F] row-major (used as B^T)
    const float*  __restrict__ bias,
    float* __restrict__ out) {      // [M][OUT_F]
  __shared__ bf16_t As[2][BM * LDT];
  __shared__ bf16_t Bs[2][BN * LDT];

  const int tid  = threadIdx.x;
  const int lane = tid & 31;
  const int wid  = tid >> 5;     // 8 waves
  const int wm   = wid >> 2;     // 0..1  -> 64-row slab
  const int wn   = wid & 3;      // 0..3  -> 32-col slab

  const int bm = blockIdx.y * BM;
  const int bn = blockIdx.x * BN;

  const v8f zero = {0.f, 0.f, 0.f, 0.f, 0.f, 0.f, 0.f, 0.f};
  v8f acc[4][2];
#pragma unroll
  for (int i = 0; i < 4; ++i)
#pragma unroll
    for (int j = 0; j < 2; ++j) acc[i][j] = zero;

  const int r  = lane & 15;
  const int kh = (lane >> 4) * 8;   // K-half select per WMMA operand layout

  // prologue: fill buffer 0
  stage_tile(X, W, As[0], Bs[0], bm, bn, 0, tid);
  wait_async0();
  __syncthreads();

  for (int k0 = 0; k0 < IN_F; k0 += BK) {
    const int buf = (k0 >> 5) & 1;

    // issue async fill of the other buffer while we compute on this one
    if (k0 + BK < IN_F)
      stage_tile(X, W, As[buf ^ 1], Bs[buf ^ 1], bm, bn, k0 + BK, tid);

    v16bf af[4], bfr[2];
#pragma unroll
    for (int mt = 0; mt < 4; ++mt)
      af[mt] = load_frag(&As[buf][(wm * 64 + mt * 16 + r) * LDT], kh);
#pragma unroll
    for (int nt = 0; nt < 2; ++nt)
      bfr[nt] = load_frag(&Bs[buf][(wn * 32 + nt * 16 + r) * LDT], kh);

#pragma unroll
    for (int mt = 0; mt < 4; ++mt)
#pragma unroll
      for (int nt = 0; nt < 2; ++nt)
        acc[mt][nt] = __builtin_amdgcn_wmma_f32_16x16x32_bf16(
            false, af[mt], false, bfr[nt], (short)0, acc[mt][nt], false, false);

    // publish the freshly staged buffer (and protect the one we just read)
    wait_async0();
    __syncthreads();
  }

  // Epilogue per C/D layout: N = lane&15, VGPR q -> M = q + 8*(lane>=16)
  const int mh = (lane >> 4) * 8;
#pragma unroll
  for (int mt = 0; mt < 4; ++mt) {
#pragma unroll
    for (int nt = 0; nt < 2; ++nt) {
      int col = bn + wn * 32 + nt * 16 + r;
      float bv = bias[col];
#pragma unroll
      for (int q = 0; q < 8; ++q) {
        int rowg = bm + wm * 64 + mt * 16 + mh + q;
        out[(size_t)rowg * OUT_F + col] = acc[mt][nt][q] + bv;
      }
    }
  }
}

extern "C" void kernel_launch(void* const* d_in, const int* in_sizes, int n_in,
                              void* d_out, int out_size, void* d_ws, size_t ws_size,
                              hipStream_t stream) {
  const float* x     = (const float*)d_in[0];   // [4,2048,4096] fp32
  const int*   codes = (const int*)d_in[1];     // [OUT*IN/2] packed bytes
  const float* scale = (const float*)d_in[2];   // [OUT,1]
  const float* bias  = (const float*)d_in[3];   // [OUT]
  float*       out   = (float*)d_out;

  const int m_tot = in_sizes[0] / IN_F;         // 8192

  char* ws = (char*)d_ws;
  bf16_t* xb = (bf16_t*)ws;                                   // 64 MB
  bf16_t* wb = (bf16_t*)(ws + (size_t)m_tot * IN_F * 2);      // +32 MB

  // pass 1a: x -> bf16 (8 elems/thread)
  {
    int threads = in_sizes[0] / 8;
    nf4_convert_x<<<threads / 256, 256, 0, stream>>>(x, xb);
  }
  // pass 1b: dequant W (4 packed ints -> 8 bf16 per thread)
  {
    int threads = in_sizes[1] / 4;
    nf4_dequant_w<<<threads / 256, 256, 0, stream>>>(codes, scale, wb);
  }
  // pass 2: WMMA GEMM
  {
    dim3 grid(OUT_F / BN, m_tot / BM);
    nf4_gemm<<<grid, 256, 0, stream>>>(xb, wb, bias, out);
  }
}
